// AMRPAAttentionWrapper_67319317397670
// MI455X (gfx1250) — matrix operand order repaired and checked by reading, not compiled, not run on hardware
//
#include <hip/hip_runtime.h>
#include <hip/hip_bf16.h>
#include <stdint.h>

// ---------------------------------------------------------------------------
// AMRPA attention, fused formulation (bf16 WMMA, fp32 accumulate):
//   q,k,v = hs @ W{q,k,v}
//   gates = sigmoid(q @ w_gate)
//   mv    = (exp(-0.5) * past_attention) @ v
//   q'    = q + gates * (mv @ Wm)            (fused into GEMM epilogue)
//   out   = softmax(q' k^T / sqrt(H)) @ v
//
// GEMM: B operands K-major (two 16B loads per fragment), A panel (32x256 bf16)
// double-buffered in LDS. Panel staging uses the Tensor Data Mover
// (tensor_load_to_lds + s_wait_tensorcnt) when the builtin is available,
// falling back to global_load_async_to_lds_b128 (ASYNCcnt) otherwise.
// B fragments double-buffered in registers; 2x4 accumulator tiles per wave.
// ---------------------------------------------------------------------------

typedef __attribute__((ext_vector_type(16))) __bf16    v16bf;
typedef __attribute__((ext_vector_type(8)))  __bf16    v8bf;
typedef __attribute__((ext_vector_type(8)))  float     v8f;
typedef __attribute__((ext_vector_type(4)))  unsigned  v4u;
typedef __attribute__((ext_vector_type(8)))  int       v8i;
typedef __attribute__((ext_vector_type(4)))  int       v4i;

#ifdef __has_builtin
#if __has_builtin(__builtin_amdgcn_tensor_load_to_lds)
#define HAVE_TDM 1
#endif
#endif
#ifndef HAVE_TDM
#define HAVE_TDM 0
#endif

__device__ __forceinline__ unsigned short f32_to_bf16_bits(float f) {
  unsigned u = __builtin_bit_cast(unsigned, f);
  u += 0x7FFFu + ((u >> 16) & 1u);            // round-to-nearest-even
  return (unsigned short)(u >> 16);
}

__device__ __forceinline__ v16bf cat8(v4u lo, v4u hi) {
  v8bf a = __builtin_bit_cast(v8bf, lo);
  v8bf b = __builtin_bit_cast(v8bf, hi);
  return __builtin_shufflevector(a, b, 0, 1, 2, 3, 4, 5, 6, 7,
                                        8, 9, 10, 11, 12, 13, 14, 15);
}

// bf16 k = 8*half..+7 and 16+8*half..+7 (table order) -> B fragment
__device__ __forceinline__ v16bf loadB16(const unsigned short* p, int half) {
  const v4u lo = *(const v4u*)(p + 8 * half);
  const v4u hi = *(const v4u*)(p + 16 + 8 * half);
  return cat8(lo, hi);
}

#if HAVE_TDM
// Issue one TDM 2-D tile load: tile_d1 x tile_d0 elements of 2 bytes from
// row-major tensor (stride0 elems/row) at gtile -> LDS[lds_byte_addr].
// Descriptor packing per CDNA5 ISA 8.3/8.4 (D# groups 0/1; groups 2/3 zero).
__device__ __forceinline__ void tdm_load_panel(const unsigned short* gtile,
                                               unsigned lds_byte_addr,
                                               unsigned tensor_d0,
                                               unsigned tensor_d1,
                                               unsigned tile_d0,
                                               unsigned tile_d1,
                                               unsigned long stride0) {
  const unsigned long ga = (unsigned long)gtile;
  v4u g0;
  g0[0] = 1u;                                          // count=1 (valid D#)
  g0[1] = lds_byte_addr;                               // lds_addr
  g0[2] = (unsigned)(ga & 0xFFFFFFFFu);                // global_addr[31:0]
  g0[3] = (unsigned)((ga >> 32) & 0x01FFFFFFu)         // global_addr[56:32]
          | (2u << 30);                                // type=2 ("image")
  v8i g1;
  g1[0] = (int)(1u << 16);                             // data_size=1 (2B)
  g1[1] = (int)((tensor_d0 & 0xFFFFu) << 16);          // tensor_dim0[15:0]
  g1[2] = (int)(((tensor_d0 >> 16) & 0xFFFFu)          // tensor_dim0[31:16]
                | ((tensor_d1 & 0xFFFFu) << 16));      // tensor_dim1[15:0]
  g1[3] = (int)(((tensor_d1 >> 16) & 0xFFFFu)          // tensor_dim1[31:16]
                | ((tile_d0 & 0xFFFFu) << 16));        // tile_dim0
  g1[4] = (int)(tile_d1 & 0xFFFFu);                    // tile_dim1; tile_dim2=0
  g1[5] = (int)(unsigned)(stride0 & 0xFFFFFFFFu);      // dim0_stride[31:0]
  g1[6] = (int)(unsigned)((stride0 >> 32) & 0xFFFFu);  // dim0_stride[47:32]
  g1[7] = 0;
  const v4i z4 = {0, 0, 0, 0};
#if __clang_major__ >= 23
  const v8i z8 = {0, 0, 0, 0, 0, 0, 0, 0};
  __builtin_amdgcn_tensor_load_to_lds(g0, g1, z4, z4, z8, 0);
#else
  __builtin_amdgcn_tensor_load_to_lds(g0, g1, z4, z4, 0);
#endif
}
#endif  // HAVE_TDM

// ---------------------------------------------------------------------------
// elementwise fp32 -> bf16 cast with scale (hs, decayed past_attention)
// ---------------------------------------------------------------------------
__global__ void __launch_bounds__(256)
cast_scale_f32_bf16(const float* __restrict__ in, unsigned short* __restrict__ out,
                    long n, float scale) {
  long i = (long)blockIdx.x * 256 + threadIdx.x;
  long stride = (long)gridDim.x * 256;
  for (; i < n; i += stride) out[i] = f32_to_bf16_bits(in[i] * scale);
}

// ---------------------------------------------------------------------------
// fused cast + transpose: fp32 in [R,Ccols] -> bf16 out [Ccols,R]
// ---------------------------------------------------------------------------
__global__ void __launch_bounds__(256)
cast_transpose_f32_bf16(const float* __restrict__ in,
                        unsigned short* __restrict__ out, int R, int Ccols) {
  __shared__ float tile[32][33];
  const int c0 = blockIdx.x * 32, r0 = blockIdx.y * 32;
  const int tx = threadIdx.x & 31, ty = threadIdx.x >> 5;   // 32 x 8
#pragma unroll
  for (int i = ty; i < 32; i += 8)
    tile[i][tx] = in[(long)(r0 + i) * Ccols + (c0 + tx)];
  __syncthreads();
#pragma unroll
  for (int i = ty; i < 32; i += 8)
    out[(long)(c0 + i) * R + (r0 + tx)] = f32_to_bf16_bits(tile[tx][i]);
}

// ---------------------------------------------------------------------------
// WMMA GEMM, B K-major:  C[M,N] = alpha * A[M,K] @ Bt[N,K]^T
//   block: 32(M) x 512(N); wave: 2x4 tiles of 16x16
//   A: LDS double-buffered panels staged by TDM (or async loads)
//   B: register double-buffered streams
// ---------------------------------------------------------------------------
#define BK 256
#define PANEL_BYTES (32 * BK * 2)      // 16 KB per buffer

__global__ void __launch_bounds__(256)
wmma_gemm_bt(const unsigned short* __restrict__ A,
             const unsigned short* __restrict__ Bt,
             float* __restrict__ C, unsigned short* __restrict__ Cb,
             int M, int N, int K,
             long sA, long sB, long sC, long sCb,
             int lda, int ldb, int ldc, int ldcb,
             int transCb, int epiGate,
             const float* __restrict__ addC,
             const float* __restrict__ rowScale,
             float alpha) {
  __shared__ unsigned short lA[2][32 * BK];   // 2 x 16 KB A panels
  const int tid  = threadIdx.x;
  const int wave = tid >> 5;
  const int lane = tid & 31;
  const int l    = lane & 15;
  const int half = lane >> 4;
  const int m0 = blockIdx.x * 32;
  const int n0 = (blockIdx.y * 8 + wave) * 64;
  const long bz = blockIdx.z;
  A  += bz * sA;
  Bt += bz * sB;

  const unsigned ldsBase = (unsigned)(uintptr_t)(&lA[0][0]);

  // stage one 32 x BK panel into buffer `buf`
  auto stage = [&](int buf, int kb) {
#if HAVE_TDM
    if (wave == 0)
      tdm_load_panel(A + (long)m0 * lda + kb,
                     ldsBase + (unsigned)buf * PANEL_BYTES,
                     (unsigned)(K - kb), (unsigned)(M - m0),
                     (unsigned)BK, 32u, (unsigned long)lda);
#else
#pragma unroll
    for (int t = 0; t < 4; ++t) {
      const int idx   = tid + t * 256;            // 0..1023
      const int row   = idx >> 5;                 // 32 x 16B chunks per row
      const int chunk = idx & 31;
      const unsigned short* g = A + (long)(m0 + row) * lda + kb + chunk * 8;
      const unsigned ldsa =
          ldsBase + (unsigned)buf * PANEL_BYTES +
          (unsigned)((row * BK + chunk * 8) * 2);
      asm volatile("global_load_async_to_lds_b128 %0, %1, off"
                   :: "v"(ldsa), "v"(g) : "memory");
    }
#endif
  };
  auto wait_stage = [&]() {
#if HAVE_TDM
    if (wave == 0) __builtin_amdgcn_s_wait_tensorcnt(0);
#else
    asm volatile("s_wait_asynccnt 0x0" ::: "memory");
#endif
    __syncthreads();
  };

  v8f acc00 = {}, acc01 = {}, acc02 = {}, acc03 = {};
  v8f acc10 = {}, acc11 = {}, acc12 = {}, acc13 = {};

  // per-lane K-major B row pointers (row n0 + 16*t + l)
  const unsigned short* bp0 = Bt + (long)(n0 +  0 + l) * ldb;
  const unsigned short* bp1 = Bt + (long)(n0 + 16 + l) * ldb;
  const unsigned short* bp2 = Bt + (long)(n0 + 32 + l) * ldb;
  const unsigned short* bp3 = Bt + (long)(n0 + 48 + l) * ldb;

  stage(0, 0);

  for (int kb = 0, it = 0; kb < K; kb += BK, ++it) {
    const int buf = it & 1;
    wait_stage();                                 // panel `buf` visible to all
    if (kb + BK < K) stage(buf ^ 1, kb + BK);     // DMA overlaps compute
    __builtin_prefetch(bp0 + kb + BK, 0, 0);      // speculative B stream hint

    const unsigned short* lbase = &lA[buf][0];

    // register double-buffered B fragments
    v16bf nb0 = loadB16(bp0 + kb, half);
    v16bf nb1 = loadB16(bp1 + kb, half);
    v16bf nb2 = loadB16(bp2 + kb, half);
    v16bf nb3 = loadB16(bp3 + kb, half);

#pragma unroll
    for (int ks = 0; ks < BK; ks += 32) {
      const v16bf b0 = nb0, b1 = nb1, b2 = nb2, b3 = nb3;
      if (ks + 32 < BK) {
        nb0 = loadB16(bp0 + kb + ks + 32, half);
        nb1 = loadB16(bp1 + kb + ks + 32, half);
        nb2 = loadB16(bp2 + kb + ks + 32, half);
        nb3 = loadB16(bp3 + kb + ks + 32, half);
      }
      // A fragments for the two 16-row tiles (ISA 7.12.2 layout)
      const v16bf a0 = cat8(*(const v4u*)&lbase[(l)      * BK + ks + 8 * half],
                            *(const v4u*)&lbase[(l)      * BK + ks + 16 + 8 * half]);
      const v16bf a1 = cat8(*(const v4u*)&lbase[(16 + l) * BK + ks + 8 * half],
                            *(const v4u*)&lbase[(16 + l) * BK + ks + 16 + 8 * half]);

      acc00 = __builtin_amdgcn_wmma_f32_16x16x32_bf16(false, a0, false, b0, (short)0, acc00, false, false);
      acc01 = __builtin_amdgcn_wmma_f32_16x16x32_bf16(false, a0, false, b1, (short)0, acc01, false, false);
      acc02 = __builtin_amdgcn_wmma_f32_16x16x32_bf16(false, a0, false, b2, (short)0, acc02, false, false);
      acc03 = __builtin_amdgcn_wmma_f32_16x16x32_bf16(false, a0, false, b3, (short)0, acc03, false, false);
      acc10 = __builtin_amdgcn_wmma_f32_16x16x32_bf16(false, a1, false, b0, (short)0, acc10, false, false);
      acc11 = __builtin_amdgcn_wmma_f32_16x16x32_bf16(false, a1, false, b1, (short)0, acc11, false, false);
      acc12 = __builtin_amdgcn_wmma_f32_16x16x32_bf16(false, a1, false, b2, (short)0, acc12, false, false);
      acc13 = __builtin_amdgcn_wmma_f32_16x16x32_bf16(false, a1, false, b3, (short)0, acc13, false, false);
    }
  }

  // ---- epilogue: VGPR r -> row m0 + 16*mt + r + 8*half, col n0 +16*t + l --
  v8f accs[2][4] = {{acc00, acc01, acc02, acc03},
                    {acc10, acc11, acc12, acc13}};
#pragma unroll
  for (int mt = 0; mt < 2; ++mt) {
#pragma unroll
    for (int t = 0; t < 4; ++t) {
#pragma unroll
      for (int r = 0; r < 8; ++r) {
        const int  m = m0 + 16 * mt + r + 8 * half;
        const int  n = n0 + 16 * t + l;
        float val = accs[mt][t][r] * alpha;
        if (epiGate)
          val = addC[bz * sC + (long)m * ldc + n] + rowScale[bz * M + m] * val;
        if (C)  C[bz * sC + (long)m * ldc + n] = val;
        if (Cb) {
          const long idx = transCb ? (sCb * bz + (long)n * ldcb + m)
                                   : (sCb * bz + (long)m * ldcb + n);
          Cb[idx] = f32_to_bf16_bits(val);
        }
      }
    }
  }
}

// ---------------------------------------------------------------------------
// gates[row] = sigmoid(dot(q[row,:], w_gate))   (one wave per row)
// ---------------------------------------------------------------------------
__global__ void __launch_bounds__(256)
gates_kernel(const float* __restrict__ q, const float* __restrict__ wg,
             float* __restrict__ g, int Hdim, long nrows) {
  const long row = (long)blockIdx.x * 8 + (threadIdx.x >> 5);
  if (row >= nrows) return;
  const int lane = threadIdx.x & 31;
  const float* qr = q + row * Hdim;
  float acc = 0.f;
  for (int i = lane; i < Hdim; i += 32) acc += qr[i] * wg[i];
  for (int off = 16; off; off >>= 1) acc += __shfl_xor(acc, off, 32);
  if (lane == 0) g[row] = 1.0f / (1.0f + __expf(-acc));
}

// ---------------------------------------------------------------------------
// row softmax: fp32 in, bf16 probs out (one 256-thread block per row)
// ---------------------------------------------------------------------------
__global__ void __launch_bounds__(256)
softmax_rows(const float* __restrict__ Smat, unsigned short* __restrict__ P,
             int ncols) {
  const long row = blockIdx.x;
  const float* s = Smat + row * (long)ncols;
  unsigned short* p = P + row * (long)ncols;
  __shared__ float red[8];
  const int tid = threadIdx.x, lane = tid & 31, wv = tid >> 5;

  float m = -3.0e38f;
  for (int i = tid; i < ncols; i += 256) m = fmaxf(m, s[i]);
  for (int off = 16; off; off >>= 1) m = fmaxf(m, __shfl_xor(m, off, 32));
  if (lane == 0) red[wv] = m;
  __syncthreads();
  float mm = red[0];
#pragma unroll
  for (int i = 1; i < 8; ++i) mm = fmaxf(mm, red[i]);
  __syncthreads();

  float acc = 0.f;
  for (int i = tid; i < ncols; i += 256) acc += __expf(s[i] - mm);
  for (int off = 16; off; off >>= 1) acc += __shfl_xor(acc, off, 32);
  if (lane == 0) red[wv] = acc;
  __syncthreads();
  float sum = 0.f;
#pragma unroll
  for (int i = 0; i < 8; ++i) sum += red[i];
  const float inv = 1.0f / sum;

  for (int i = tid; i < ncols; i += 256)
    p[i] = f32_to_bf16_bits(__expf(s[i] - mm) * inv);
}

// ---------------------------------------------------------------------------
// launcher
// ---------------------------------------------------------------------------
extern "C" void kernel_launch(void* const* d_in, const int* in_sizes, int n_in,
                              void* d_out, int out_size, void* d_ws, size_t ws_size,
                              hipStream_t stream) {
  (void)in_sizes; (void)n_in; (void)out_size; (void)ws_size;
  constexpr int  Bc = 4, Sc = 2048, Hc = 1024;
  constexpr long BSH = (long)Bc * Sc * Hc;
  constexpr long BSS = (long)Bc * Sc * Sc;
  constexpr long HH  = (long)Hc * Hc;

  const float* hs = (const float*)d_in[0];
  const float* pa = (const float*)d_in[1];
  const float* Wq = (const float*)d_in[2];
  const float* Wk = (const float*)d_in[3];
  const float* Wv = (const float*)d_in[4];
  const float* Wm = (const float*)d_in[5];
  const float* wg = (const float*)d_in[6];
  float* out = (float*)d_out;

  char* w = (char*)d_ws;
  auto alloc = [&](size_t bytes) -> void* {
    void* p = (void*)w;
    w += (bytes + 255) & ~(size_t)255;
    return p;
  };
  unsigned short* hs_bf  = (unsigned short*)alloc(BSH * 2);
  unsigned short* wqT    = (unsigned short*)alloc(HH * 2);
  unsigned short* wkT    = (unsigned short*)alloc(HH * 2);
  unsigned short* wvT    = (unsigned short*)alloc(HH * 2);
  unsigned short* wmT    = (unsigned short*)alloc(HH * 2);
  unsigned short* mem_bf = (unsigned short*)alloc(BSS * 2);
  float*          q_f    = (float*)alloc(BSH * 4);
  unsigned short* k_bf   = (unsigned short*)alloc(BSH * 2);   // [B,S,H] K-major for scores
  unsigned short* vT_bf  = (unsigned short*)alloc(BSH * 2);   // [B,H,S] K-major for mv/context
  unsigned short* mv_bf  = (unsigned short*)alloc(BSH * 2);
  float*          gates  = (float*)alloc((long)Bc * Sc * 4);
  unsigned short* qp_bf  = (unsigned short*)alloc(BSH * 2);
  float*          scores = (float*)alloc(BSS * 4);
  unsigned short* pr_bf  = (unsigned short*)alloc(BSS * 2);

  const float decay      = 0.60653065971263342f;  // exp(-0.5)
  const float inv_sqrt_h = 0.03125f;              // 1/sqrt(1024)
  const long  SH = (long)Sc * Hc, SS = (long)Sc * Sc, HS = (long)Hc * Sc;

  // 1. casts / transposed weight casts
  cast_scale_f32_bf16<<<4096, 256, 0, stream>>>(hs, hs_bf, BSH, 1.0f);
  cast_scale_f32_bf16<<<8192, 256, 0, stream>>>(pa, mem_bf, BSS, decay);
  dim3 g_tr(Hc / 32, Hc / 32);
  cast_transpose_f32_bf16<<<g_tr, 256, 0, stream>>>(Wq, wqT, Hc, Hc);
  cast_transpose_f32_bf16<<<g_tr, 256, 0, stream>>>(Wk, wkT, Hc, Hc);
  cast_transpose_f32_bf16<<<g_tr, 256, 0, stream>>>(Wv, wvT, Hc, Hc);
  cast_transpose_f32_bf16<<<g_tr, 256, 0, stream>>>(Wm, wmT, Hc, Hc);

  // 2. projections (batched over B)
  dim3 g_proj(Sc / 32, Hc / 512, Bc);
  wmma_gemm_bt<<<g_proj, 256, 0, stream>>>(hs_bf, wqT, q_f, nullptr,
      Sc, Hc, Hc, SH, 0, SH, 0, Hc, Hc, Hc, 0, 0, 0, nullptr, nullptr, 1.0f);
  wmma_gemm_bt<<<g_proj, 256, 0, stream>>>(hs_bf, wkT, nullptr, k_bf,
      Sc, Hc, Hc, SH, 0, 0, SH, Hc, Hc, Hc, Hc, 0, 0, nullptr, nullptr, 1.0f);
  wmma_gemm_bt<<<g_proj, 256, 0, stream>>>(hs_bf, wvT, nullptr, vT_bf,
      Sc, Hc, Hc, SH, 0, 0, SH, Hc, Hc, Hc, Sc, 1, 0, nullptr, nullptr, 1.0f);

  // 3. gates = sigmoid(q @ w_gate)
  gates_kernel<<<(Bc * Sc) / 8, 256, 0, stream>>>(q_f, wg, gates, Hc,
                                                  (long)Bc * Sc);

  // 4. mv = memory @ v   (batched [S,S] @ vT[H,S]^T)
  wmma_gemm_bt<<<g_proj, 256, 0, stream>>>(mem_bf, vT_bf, nullptr, mv_bf,
      Sc, Hc, Sc, SS, HS, 0, SH, Sc, Sc, Hc, Hc, 0, 0, nullptr, nullptr, 1.0f);

  // 5. q' = q + gates * (mv @ Wm)   (flat M = B*S, fused epilogue)
  dim3 g_t(Bc * Sc / 32, Hc / 512, 1);
  wmma_gemm_bt<<<g_t, 256, 0, stream>>>(mv_bf, wmT, nullptr, qp_bf,
      Bc * Sc, Hc, Hc, 0, 0, 0, 0, Hc, Hc, Hc, Hc, 0, 1, q_f, gates, 1.0f);

  // 6. scores = q' @ k^T * inv_sqrt_h  (batched; k_bf already K-major)
  dim3 g_sc(Sc / 32, Sc / 512, Bc);
  wmma_gemm_bt<<<g_sc, 256, 0, stream>>>(qp_bf, k_bf, scores, nullptr,
      Sc, Sc, Hc, SH, SH, SS, 0, Hc, Hc, Sc, Sc, 0, 0, nullptr, nullptr,
      inv_sqrt_h);

  // 7. probs = softmax(scores)
  softmax_rows<<<Bc * Sc, 256, 0, stream>>>(scores, pr_bf, Sc);

  // 8. context = probs @ v  (batched [S,S] @ vT[H,S]^T) -> d_out
  wmma_gemm_bt<<<g_proj, 256, 0, stream>>>(pr_bf, vT_bf, out, nullptr,
      Sc, Hc, Sc, SS, HS, SH, 0, Sc, Sc, Hc, Hc, 0, 0, nullptr, nullptr, 1.0f);
}
